// SimpleNet_28321014350382
// MI455X (gfx1250) — matrix operand order repaired and checked by reading, not compiled, not run on hardware
//
#include <hip/hip_runtime.h>
#include <hip/hip_bf16.h>
#include <stdint.h>

// ---------------------------------------------------------------------------
// GNN (GIN-ish) forward for MI455X / gfx1250.
//   * bf16 WMMA (v_wmma_f32_16x16x32_bf16) for every H=128-wide GEMM
//   * edge encoder computed ONCE (conv1 params reused 4x in the reference)
//   * BN folded to per-channel affine; batch stats accumulated in GEMM epilogue
//   * message pass: 1 wave32 per edge, f32 global atomics for segment_sum
//
// Input flattening assumption (compile-only environment, cannot verify):
//   d_in[0..5] = var_feat, con_feat, edge_feat, edge_index, assoc_var, assoc_con
//   d_in[6..]  = params leaves, jax tree order (dict keys alphabetical):
//     con_enc{b1,b2,w1,w2}, conv1{edge_encoder{b1,b2,beta,gamma,w1,w2}, eps,
//     mlp{b1,b2,beta,gamma,w1,w2}}, lin1(w,b)..lin4(w,b), var_enc{b1,b2,w1,w2}
// ---------------------------------------------------------------------------

#define H 128
#define NVAR 60000
#define NCON 40000
#define NNODES (NVAR + NCON)
#define NEDGES 1000000
#define BN_EPS 1e-5f

typedef __attribute__((ext_vector_type(16))) __bf16 v16bf;
typedef __attribute__((ext_vector_type(8))) float v8f;

union AFrag {
  v16bf v;
  unsigned short u[16];
  unsigned int d[8];
  uint4 q[2];
};

__device__ __forceinline__ unsigned short f2bf(float f) {
  unsigned int u = __float_as_uint(f);
  u += 0x7FFFu + ((u >> 16) & 1u);   // round-to-nearest-even
  return (unsigned short)(u >> 16);
}
__device__ __forceinline__ float bf2f(unsigned short b) {
  return __uint_as_float(((unsigned int)b) << 16);
}
__device__ __forceinline__ void atomAddF(float* p, float v) {
  unsafeAtomicAdd(p, v);  // global_atomic_add_f32 / ds_add_f32
}

// ---------------------------------------------------------------------------
// Pack an f32 [K,128] row-major weight into the bf16 B-fragment layout.
// B (KxN, K chunk of 32, N tile of 16): lane = N%16 (+half picks K 8-block in each 16),
// lane-local ushort j  <->  K = kc*32 + 8*half + j (j<8), kc*32 + 16 + 8*half + (j-8) (j>=8).  8 packed dwords / lane.
// blockIdx.x = kc*8 + tile, 32 threads.
// ---------------------------------------------------------------------------
__global__ void pack_weight_kernel(const float* __restrict__ W,
                                   unsigned int* __restrict__ Wp) {
  int tile = blockIdx.x & 7;
  int kc = blockIdx.x >> 3;
  int l = threadIdx.x;
  int n = tile * 16 + (l & 15);
  int half = l >> 4;
  unsigned int* dst = Wp + ((size_t)blockIdx.x * 32 + l) * 8;
#pragma unroll
  for (int v = 0; v < 8; ++v) {
    int k0 = kc * 32 + 8 * half + 2 * v + (v >= 4 ? 8 : 0);
    unsigned int lo = f2bf(W[(size_t)k0 * H + n]);
    unsigned int hi = f2bf(W[(size_t)(k0 + 1) * H + n]);
    dst[v] = lo | (hi << 16);
  }
}

// ---------------------------------------------------------------------------
// WMMA GEMM: out[M,128] = act(A[M,K] @ W + bias), A in bf16 (or synthesized).
// MODE 0: A = bf16 buffer [M,K]
// MODE 1: A = relu(w1[k]*f[row] + b1[k])      (edge encoder first linear, K=128)
// MODE 2: A = (1+eps)*x[row,k] + agg[row,k]   (GIN combine, K=128)
// MODE 3: A = concat(x0..x4)[row,k]           (head lin1, K=640)
// RELU: apply relu.  STATS: accumulate per-channel sum/sumsq for BN.
// ---------------------------------------------------------------------------
struct GemmArgs {
  const unsigned short* A;
  const unsigned short* x0;
  const unsigned short* x1;
  const unsigned short* x2;
  const unsigned short* x3;
  const unsigned short* x4;
  const float* aggf;
  const float* edgef;
  const float* w1;
  const float* b1;
  const float* epsp;
  const unsigned int* Wp;
  const float* bias;
  unsigned short* out;
  float* stats;
  int M;
  int K;
};

template <int MODE, int RELU, int STATS>
__global__ __launch_bounds__(256) void gemm_n128(GemmArgs g) {
  __shared__ float ssum[2][H];
  const int lane = threadIdx.x & 31;
  const int wave = threadIdx.x >> 5;
  const int half = lane >> 4;
  const int mloc = lane & 15;
  const int rowBase = blockIdx.x * 128 + wave * 16;
  int arow = rowBase + mloc;
  if (arow >= g.M) arow = g.M - 1;  // clamp; OOB rows never stored/counted

  if (STATS) {
    if (threadIdx.x < H) { ssum[0][threadIdx.x] = 0.f; ssum[1][threadIdx.x] = 0.f; }
    __syncthreads();
  }

  v8f acc[8];
#pragma unroll
  for (int t = 0; t < 8; ++t)
#pragma unroll
    for (int r = 0; r < 8; ++r) acc[t][r] = 0.f;

  float fval = 0.f, hscale = 1.f;
  if (MODE == 1) fval = g.edgef[arow];
  if (MODE == 2) hscale = 1.f + g.epsp[0];

  const int nkc = g.K >> 5;
  for (int kc = 0; kc < nkc; ++kc) {
    // ---- A fragment: per-lane row = arow, K ushorts 0..7 -> 8*half+(0..7),
    //                  ushorts 8..15 -> 16+8*half+(0..7) (ISA 16-bit A layout)
    AFrag a;
    if (MODE == 0) {
      const unsigned short* rp = g.A + (size_t)arow * g.K + kc * 32 + 8 * half;
      a.q[0] = *(const uint4*)rp;
      a.q[1] = *(const uint4*)(rp + 16);
    } else if (MODE == 1) {
      int kb = kc * 32 + 8 * half;
#pragma unroll
      for (int j = 0; j < 8; ++j) {
        int k = kb + j;
        a.u[j] = f2bf(fmaxf(fmaf(g.w1[k], fval, g.b1[k]), 0.f));
      }
#pragma unroll
      for (int j = 0; j < 8; ++j) {
        int k = kb + 16 + j;
        a.u[8 + j] = f2bf(fmaxf(fmaf(g.w1[k], fval, g.b1[k]), 0.f));
      }
    } else if (MODE == 2) {
      const unsigned short* rp = g.A + (size_t)arow * H + kc * 32 + 8 * half;
      const float* ap = g.aggf + (size_t)arow * H + kc * 32 + 8 * half;
#pragma unroll
      for (int j = 0; j < 8; ++j) a.u[j] = f2bf(fmaf(hscale, bf2f(rp[j]), ap[j]));
#pragma unroll
      for (int j = 0; j < 8; ++j) a.u[8 + j] = f2bf(fmaf(hscale, bf2f(rp[16 + j]), ap[16 + j]));
    } else {  // MODE == 3: concat of five [M,128] bf16 buffers, K=640
      int sel = kc >> 2;
      const unsigned short* bp = sel == 0   ? g.x0
                                 : sel == 1 ? g.x1
                                 : sel == 2 ? g.x2
                                 : sel == 3 ? g.x3
                                            : g.x4;
      const unsigned short* rp = bp + (size_t)arow * H + (kc & 3) * 32 + 8 * half;
      a.q[0] = *(const uint4*)rp;
      a.q[1] = *(const uint4*)(rp + 16);
    }

    // ---- 8 column tiles, pre-packed B fragments, one WMMA each
    const unsigned int* wp = g.Wp + ((size_t)kc * 8 * 32 + lane) * 8;
#pragma unroll
    for (int t = 0; t < 8; ++t) {
      AFrag b;
      const unsigned int* bp = wp + (size_t)t * 32 * 8;
      b.q[0] = *(const uint4*)bp;
      b.q[1] = *(const uint4*)(bp + 4);
      acc[t] = __builtin_amdgcn_wmma_f32_16x16x32_bf16(false, a.v, false, b.v,
                                                       (short)0, acc[t], false, false);
    }
  }

  // ---- epilogue: bias (+relu), bf16 store, optional BN stats
#pragma unroll
  for (int t = 0; t < 8; ++t) {
    int col = t * 16 + mloc;
    float bv = g.bias[col];
    float s = 0.f, sq = 0.f;
#pragma unroll
    for (int r = 0; r < 8; ++r) {
      int row = rowBase + r + 8 * half;  // C/D layout: VGPR r -> M = r + 8*half
      float v = acc[t][r] + bv;
      if (RELU) v = fmaxf(v, 0.f);
      if (row < g.M) {
        g.out[(size_t)row * H + col] = f2bf(v);
        if (STATS) { s += v; sq += v * v; }
      }
    }
    if (STATS) {
      atomAddF(&ssum[0][col], s);
      atomAddF(&ssum[1][col], sq);
    }
  }
  if (STATS) {
    __syncthreads();
    if (threadIdx.x < H) {
      atomAddF(g.stats + threadIdx.x, ssum[0][threadIdx.x]);
      atomAddF(g.stats + H + threadIdx.x, ssum[1][threadIdx.x]);
    }
  }
}

// ---------------------------------------------------------------------------
__global__ void zero_f32_kernel(float* __restrict__ p, long long n) {
  long long i = (long long)blockIdx.x * blockDim.x + threadIdx.x;
  if (i < n) p[i] = 0.f;
}

// node encoder first linear: [M,3] @ [3,128] + b, relu -> bf16 [M,128]
__global__ void enc_l1_kernel(const float* __restrict__ feat, const float* __restrict__ w1,
                              const float* __restrict__ b1, unsigned short* __restrict__ out,
                              int M) {
  int idx = blockIdx.x * blockDim.x + threadIdx.x;
  if (idx >= M * H) return;
  int row = idx >> 7, c = idx & (H - 1);
  float v = b1[c];
  v = fmaf(feat[row * 3 + 0], w1[0 * H + c], v);
  v = fmaf(feat[row * 3 + 1], w1[1 * H + c], v);
  v = fmaf(feat[row * 3 + 2], w1[2 * H + c], v);
  out[idx] = f2bf(fmaxf(v, 0.f));
}

// BN: stats -> per-channel affine  y = a*x + c
__global__ void bn_finalize_kernel(const float* __restrict__ stats, const float* __restrict__ gamma,
                                   const float* __restrict__ beta, float* __restrict__ ab,
                                   float invCount) {
  int c = threadIdx.x;
  float m = stats[c] * invCount;
  float var = stats[H + c] * invCount - m * m;
  float a = gamma[c] * rsqrtf(var + BN_EPS);
  ab[c] = a;
  ab[H + c] = fmaf(-a, m, beta[c]);
}

// x_{l+1} = relu(a*u + c)
__global__ void bn_apply_relu_kernel(const unsigned short* __restrict__ u,
                                     const float* __restrict__ ab,
                                     unsigned short* __restrict__ xn, long long total) {
  long long i = (long long)blockIdx.x * blockDim.x + threadIdx.x;
  if (i >= total) return;
  int c = (int)(i & (H - 1));
  float v = fmaf(ab[c], bf2f(u[i]), ab[H + c]);
  xn[i] = f2bf(fmaxf(v, 0.f));
}

// msg = relu(x[src] + bn(e)); agg[dst] += msg.  One wave per edge, 4 ch/lane.
__global__ void message_scatter_kernel(const unsigned short* __restrict__ x,
                                       const unsigned short* __restrict__ e,
                                       const float* __restrict__ eab,
                                       const int* __restrict__ src, const int* __restrict__ dst,
                                       float* __restrict__ agg) {
  int t = blockIdx.x * blockDim.x + threadIdx.x;
  int edge = t >> 5;
  if (edge >= NEDGES) return;
  int lane = t & 31;
  int s = src[edge], d = dst[edge];
  int c0 = lane * 4;
  const unsigned short* xp = x + (size_t)s * H + c0;
  const unsigned short* ep = e + (size_t)edge * H + c0;
  float* ap = agg + (size_t)d * H + c0;
  ushort4 xv = *(const ushort4*)xp;
  ushort4 ev = *(const ushort4*)ep;
  float m0 = fmaxf(bf2f(xv.x) + fmaf(eab[c0 + 0], bf2f(ev.x), eab[H + c0 + 0]), 0.f);
  float m1 = fmaxf(bf2f(xv.y) + fmaf(eab[c0 + 1], bf2f(ev.y), eab[H + c0 + 1]), 0.f);
  float m2 = fmaxf(bf2f(xv.z) + fmaf(eab[c0 + 2], bf2f(ev.z), eab[H + c0 + 2]), 0.f);
  float m3 = fmaxf(bf2f(xv.w) + fmaf(eab[c0 + 3], bf2f(ev.w), eab[H + c0 + 3]), 0.f);
  atomAddF(ap + 0, m0);
  atomAddF(ap + 1, m1);
  atomAddF(ap + 2, m2);
  atomAddF(ap + 3, m3);
}

// lin4 [128,2] + log_softmax
__global__ void head_final_kernel(const unsigned short* __restrict__ h3,
                                  const float* __restrict__ w4, const float* __restrict__ b4,
                                  float* __restrict__ out, int M) {
  int row = blockIdx.x * blockDim.x + threadIdx.x;
  if (row >= M) return;
  float o0 = b4[0], o1 = b4[1];
  const unsigned short* hp = h3 + (size_t)row * H;
#pragma unroll 8
  for (int k = 0; k < H; ++k) {
    float h = bf2f(hp[k]);
    o0 = fmaf(h, w4[k * 2 + 0], o0);
    o1 = fmaf(h, w4[k * 2 + 1], o1);
  }
  float mx = fmaxf(o0, o1);
  float lse = mx + logf(expf(o0 - mx) + expf(o1 - mx));
  out[row * 2 + 0] = o0 - lse;
  out[row * 2 + 1] = o1 - lse;
}

// ---------------------------------------------------------------------------
extern "C" void kernel_launch(void* const* d_in, const int* in_sizes, int n_in,
                              void* d_out, int out_size, void* d_ws, size_t ws_size,
                              hipStream_t stream) {
  (void)in_sizes; (void)n_in; (void)out_size; (void)ws_size;

  const float* varf = (const float*)d_in[0];
  const float* conf = (const float*)d_in[1];
  const float* edgef = (const float*)d_in[2];
  const int* eidx = (const int*)d_in[3];
  // d_in[4], d_in[5]: assoc arrays == identity layout (var rows 0..NVAR, con after)

  enum {
    P = 6,
    CON_B1 = P + 0, CON_B2, CON_W1, CON_W2,
    EE_B1, EE_B2, EE_BETA, EE_GAMMA, EE_W1, EE_W2,
    EPS,
    MLP_B1, MLP_B2, MLP_BETA, MLP_GAMMA, MLP_W1, MLP_W2,
    LIN1_W, LIN1_B, LIN2_W, LIN2_B, LIN3_W, LIN3_B, LIN4_W, LIN4_B,
    VAR_B1, VAR_B2, VAR_W1, VAR_W2
  };
  auto F = [&](int i) { return (const float*)d_in[i]; };

  // ---- workspace carve-up (256B aligned) ----
  char* ws = (char*)d_ws;
  size_t off = 0;
  auto alloc = [&](size_t bytes) -> void* {
    void* p = ws + off;
    off += (bytes + 255) & ~(size_t)255;
    return p;
  };
  unsigned int* wp_var2 = (unsigned int*)alloc((size_t)H * H * 2);
  unsigned int* wp_con2 = (unsigned int*)alloc((size_t)H * H * 2);
  unsigned int* wp_ee2 = (unsigned int*)alloc((size_t)H * H * 2);
  unsigned int* wp_mlp1 = (unsigned int*)alloc((size_t)H * H * 2);
  unsigned int* wp_mlp2 = (unsigned int*)alloc((size_t)H * H * 2);
  unsigned int* wp_l1 = (unsigned int*)alloc((size_t)5 * H * H * 2);
  unsigned int* wp_l2 = (unsigned int*)alloc((size_t)H * H * 2);
  unsigned int* wp_l3 = (unsigned int*)alloc((size_t)H * H * 2);
  unsigned short* xs[5];
  for (int i = 0; i < 5; ++i) xs[i] = (unsigned short*)alloc((size_t)NNODES * H * 2);
  unsigned short* ebuf = (unsigned short*)alloc((size_t)NEDGES * H * 2);
  float* agg = (float*)alloc((size_t)NNODES * H * 4);
  unsigned short* tmpA = (unsigned short*)alloc((size_t)NNODES * H * 2);
  unsigned short* ubuf = (unsigned short*)alloc((size_t)NNODES * H * 2);
  float* estats = (float*)alloc(2 * H * 4);
  float* eab = (float*)alloc(2 * H * 4);
  float* nstats = (float*)alloc(2 * H * 4);
  float* nab = (float*)alloc(2 * H * 4);

  // ---- pack weights into WMMA B-fragment layout ----
  auto pack = [&](int leaf, unsigned int* dst, int K) {
    pack_weight_kernel<<<(K / 32) * 8, 32, 0, stream>>>(F(leaf), dst);
  };
  pack(VAR_W2, wp_var2, H);
  pack(CON_W2, wp_con2, H);
  pack(EE_W2, wp_ee2, H);
  pack(MLP_W1, wp_mlp1, H);
  pack(MLP_W2, wp_mlp2, H);
  pack(LIN1_W, wp_l1, 5 * H);
  pack(LIN2_W, wp_l2, H);
  pack(LIN3_W, wp_l3, H);

  // ---- node encoders -> xs[0] ----
  enc_l1_kernel<<<(NVAR * H + 255) / 256, 256, 0, stream>>>(varf, F(VAR_W1), F(VAR_B1), tmpA, NVAR);
  {
    GemmArgs ga{};
    ga.A = tmpA; ga.Wp = wp_var2; ga.bias = F(VAR_B2);
    ga.out = xs[0]; ga.M = NVAR; ga.K = H;
    gemm_n128<0, 0, 0><<<(NVAR + 127) / 128, 256, 0, stream>>>(ga);
  }
  enc_l1_kernel<<<(NCON * H + 255) / 256, 256, 0, stream>>>(conf, F(CON_W1), F(CON_B1), tmpA, NCON);
  {
    GemmArgs ga{};
    ga.A = tmpA; ga.Wp = wp_con2; ga.bias = F(CON_B2);
    ga.out = xs[0] + (size_t)NVAR * H; ga.M = NCON; ga.K = H;
    gemm_n128<0, 0, 0><<<(NCON + 127) / 128, 256, 0, stream>>>(ga);
  }

  // ---- edge encoder, computed once (conv1 reused 4x in reference) ----
  zero_f32_kernel<<<1, 256, 0, stream>>>(estats, 2 * H);
  {
    GemmArgs ga{};
    ga.edgef = edgef; ga.w1 = F(EE_W1); ga.b1 = F(EE_B1);
    ga.Wp = wp_ee2; ga.bias = F(EE_B2);
    ga.out = ebuf; ga.stats = estats; ga.M = NEDGES; ga.K = H;
    gemm_n128<1, 1, 1><<<(NEDGES + 127) / 128, 256, 0, stream>>>(ga);
  }
  bn_finalize_kernel<<<1, H, 0, stream>>>(estats, F(EE_GAMMA), F(EE_BETA), eab, 1.f / NEDGES);

  const int* srcp = eidx;
  const int* dstp = eidx + NEDGES;
  const long long nodeElems = (long long)NNODES * H;

  // ---- 4 message-passing layers ----
  for (int l = 0; l < 4; ++l) {
    zero_f32_kernel<<<(int)((nodeElems + 255) / 256), 256, 0, stream>>>(agg, nodeElems);
    message_scatter_kernel<<<(NEDGES * 32 + 255) / 256, 256, 0, stream>>>(xs[l], ebuf, eab, srcp, dstp, agg);
    {
      GemmArgs ga{};
      ga.A = xs[l]; ga.aggf = agg; ga.epsp = F(EPS);
      ga.Wp = wp_mlp1; ga.bias = F(MLP_B1);
      ga.out = tmpA; ga.M = NNODES; ga.K = H;
      gemm_n128<2, 1, 0><<<(NNODES + 127) / 128, 256, 0, stream>>>(ga);
    }
    zero_f32_kernel<<<1, 256, 0, stream>>>(nstats, 2 * H);
    {
      GemmArgs ga{};
      ga.A = tmpA; ga.Wp = wp_mlp2; ga.bias = F(MLP_B2);
      ga.out = ubuf; ga.stats = nstats; ga.M = NNODES; ga.K = H;
      gemm_n128<0, 1, 1><<<(NNODES + 127) / 128, 256, 0, stream>>>(ga);
    }
    bn_finalize_kernel<<<1, H, 0, stream>>>(nstats, F(MLP_GAMMA), F(MLP_BETA), nab, 1.f / NNODES);
    bn_apply_relu_kernel<<<(int)((nodeElems + 255) / 256), 256, 0, stream>>>(ubuf, nab, xs[l + 1], nodeElems);
  }

  // ---- head: concat(5H) -> 128 -> 128 -> 128 -> 2 -> log_softmax ----
  {
    GemmArgs ga{};
    ga.x0 = xs[0]; ga.x1 = xs[1]; ga.x2 = xs[2]; ga.x3 = xs[3]; ga.x4 = xs[4];
    ga.Wp = wp_l1; ga.bias = F(LIN1_B);
    ga.out = tmpA; ga.M = NVAR; ga.K = 5 * H;
    gemm_n128<3, 1, 0><<<(NVAR + 127) / 128, 256, 0, stream>>>(ga);
  }
  {
    GemmArgs ga{};
    ga.A = tmpA; ga.Wp = wp_l2; ga.bias = F(LIN2_B);
    ga.out = ubuf; ga.M = NVAR; ga.K = H;
    gemm_n128<0, 1, 0><<<(NVAR + 127) / 128, 256, 0, stream>>>(ga);
    ga.A = ubuf; ga.Wp = wp_l3; ga.bias = F(LIN3_B); ga.out = tmpA;
    gemm_n128<0, 1, 0><<<(NVAR + 127) / 128, 256, 0, stream>>>(ga);
  }
  head_final_kernel<<<(NVAR + 255) / 256, 256, 0, stream>>>(tmpA, F(LIN4_W), F(LIN4_B), (float*)d_out, NVAR);
}
